// GraphAttentionLayer_36077725286420
// MI455X (gfx1250) — matrix operand order repaired and checked by reading, not compiled, not run on hardware
//
#include <hip/hip_runtime.h>

// GAT layer, fused attention, MI455X (gfx1250) wave32 + WMMA f16->f32.
// Shapes fixed by the reference: B=8, N=2048, F_in=256, F_out=128.

#define LRELU_ALPHA 0.2f
#define BB   8
#define NN   2048
#define FIN  256
#define FOUT 128
#define SEGS 8          // row segments for the column-stat pass

typedef __attribute__((ext_vector_type(16))) _Float16 v16h;
typedef __attribute__((ext_vector_type(8)))  _Float16 v8h;
typedef __attribute__((ext_vector_type(8)))  float    v8f;

__device__ __forceinline__ float leaky(float v) { return v > 0.f ? v : LRELU_ALPHA * v; }

// Assemble the 16x32 f16 A operand from a row-major LDS tile with a given row
// stride, contiguous K layout: lane<16 -> M=lane, K = {0..15};
// lane>=16 -> M=lane-16, K = {16..31}. kbase selects a K=32 sub-chunk.
__device__ __forceinline__ v16h load_A_from_lds(const _Float16* tile, int stride,
                                                int kbase, int lane) {
  int row = lane & 15;
  int ka  = kbase + ((lane < 16) ? 0 : 16);
  v8h lo = *(const v8h*)(tile + row * stride + ka);
  v8h hi = *(const v8h*)(tile + row * stride + ka + 8);
  v16h a;
#pragma unroll
  for (int p = 0; p < 8; ++p) { a[p] = lo[p]; a[p + 8] = hi[p]; }
  return a;
}

// ---------------------------------------------------------------------------
// Kernel 1: h = x @ W^T  (M=B*N, K=FIN, O=FOUT), written transposed as
// hT[b][o][m] in f16 so the attention GEMM's B operand is contiguous per lane.
// Block = 256 threads (8 waves). Each block: one 16-row m-tile x all 128 cols.
// ---------------------------------------------------------------------------
__global__ __launch_bounds__(256)
void k_h_gemm(const float* __restrict__ x, const float* __restrict__ W,
              _Float16* __restrict__ hT) {
  __shared__ __align__(16) _Float16 sA[16 * 32];
  __shared__ __align__(16) _Float16 sC[FOUT * 16];

  int blk  = blockIdx.x;            // B * N/16 = 1024 blocks
  int b    = blk >> 7;
  int m0   = (blk & 127) << 4;
  int t    = threadIdx.x;
  int lane = t & 31;
  int wave = t >> 5;
  int nw   = wave << 4;             // this wave's output-column base

  // staging indices (constant across K loop)
  int sidx = t * 2;
  int srow = sidx >> 5, scol = sidx & 31;
  const float* xrow = x + ((size_t)(b * NN + m0 + srow) * FIN) + scol;

  v8f c = {};
  for (int k0 = 0; k0 < FIN; k0 += 32) {
    // cooperative stage: x[b, m0:m0+16, k0:k0+32] f32 -> f16 in LDS
    sA[srow * 32 + scol]     = (_Float16)xrow[k0];
    sA[srow * 32 + scol + 1] = (_Float16)xrow[k0 + 1];
    __syncthreads();

    v16h a = load_A_from_lds(sA, 32, 0, lane);

    // B operand: W rows are contiguous in k. lane -> col o = nw + lane%16,
    // K range (lane<16 ? k0..k0+15 : k0+16..k0+31).
    int o  = nw + (lane & 15);
    int kk = k0 + ((lane < 16) ? 0 : 16);
    const float* wp = W + (size_t)o * FIN + kk;
    v16h bm;
#pragma unroll
    for (int p = 0; p < 16; ++p) bm[p] = (_Float16)wp[p];

    c = __builtin_amdgcn_wmma_f32_16x16x32_f16(false, a, false, bm,
                                               (short)0, c, false, false);
    __syncthreads();
  }

  // transpose C via LDS and store hT[b][o][m0:m0+16] (contiguous 32B rows)
  int nl = lane & 15;
  int mh = (lane < 16) ? 0 : 8;
#pragma unroll
  for (int r = 0; r < 8; ++r)
    sC[(nw + nl) * 16 + r + mh] = (_Float16)c[r];
  __syncthreads();

  if (t < FOUT) {
    const float4* s = (const float4*)(sC + t * 16);
    float4* d = (float4*)(hT + ((size_t)(b * FOUT + t) * NN + m0));
    d[0] = s[0];
    d[1] = s[1];
  }
}

// ---------------------------------------------------------------------------
// Kernel 2: s1[b,j] = sum_o hT[b,o,j]*a1[o];  s2 likewise. Coalesced: fixed o,
// consecutive threads read consecutive j.
// ---------------------------------------------------------------------------
__global__ __launch_bounds__(256)
void k_rowvec(const _Float16* __restrict__ hT, const float* __restrict__ a1,
              const float* __restrict__ a2, float* __restrict__ s1,
              float* __restrict__ s2) {
  int jg = blockIdx.x * 256 + threadIdx.x;    // 0 .. B*N-1
  int b  = jg >> 11;
  int j  = jg & (NN - 1);
  const _Float16* hp = hT + (size_t)(b * FOUT) * NN + j;
  float acc1 = 0.f, acc2 = 0.f;
#pragma unroll 4
  for (int o = 0; o < FOUT; ++o) {
    float hv = (float)hp[(size_t)o * NN];
    acc1 += hv * a1[o];
    acc2 += hv * a2[o];
  }
  s1[jg] = acc1;
  s2[jg] = acc2;
}

// ---------------------------------------------------------------------------
// Kernel 3a: per-column (axis=1) online softmax partials over a row segment.
// One thread per column j; block covers 256 consecutive j (coalesced adj rows).
// ---------------------------------------------------------------------------
__global__ __launch_bounds__(256)
void k_colstats_partial(const float* __restrict__ adj, const float* __restrict__ s1,
                        const float* __restrict__ s2, const float* __restrict__ ab,
                        float* __restrict__ pm, float* __restrict__ ps) {
  int chunk = blockIdx.x;                 // B*N/256 = 64
  int seg   = blockIdx.y;                 // SEGS
  int t     = threadIdx.x;
  int jg    = chunk * 256 + t;
  int b     = jg >> 11;
  float s2j = s2[jg] + ab[0];
  const float* s1b = s1 + (b << 11);
  const float* ap  = adj + (size_t)(b * NN) * NN + (jg & (NN - 1));

  float m = -3.0e38f, s = 0.f;
  int i0 = seg * (NN / SEGS);
  for (int i = i0; i < i0 + NN / SEGS; ++i) {
    float a = ap[(size_t)i * NN];
    if (a > 0.f) {
      float e = leaky(s1b[i] + s2j);
      if (e > m) { s = s * __expf(m - e) + 1.f; m = e; }
      else        s += __expf(e - m);
    }
  }
  pm[seg * (BB * NN) + jg] = m;
  ps[seg * (BB * NN) + jg] = s;
}

// Kernel 3b: merge segment partials into a single fold z = max + log(denom),
// so the attention kernel computes p = exp(e - z) with one table lookup.
// (diag guarantees >=1 unmasked entry per column, so denom > 0).
__global__ __launch_bounds__(256)
void k_colstats_merge(const float* __restrict__ pm, const float* __restrict__ ps,
                      float* __restrict__ zcol) {
  int jg = blockIdx.x * 256 + threadIdx.x;
  float m = -3.0e38f;
#pragma unroll
  for (int s = 0; s < SEGS; ++s) m = fmaxf(m, pm[s * (BB * NN) + jg]);
  float d = 0.f;
#pragma unroll
  for (int s = 0; s < SEGS; ++s)
    d += ps[s * (BB * NN) + jg] * __expf(pm[s * (BB * NN) + jg] - m);
  zcol[jg] = m + __logf(d);
}

// ---------------------------------------------------------------------------
// Kernel 4: h'[b,i,:] = sum_j P[i,j] * h[j,:] with P built on the fly from adj
// + folded softmax stats. Block = 8 waves sharing one 16-row i-tile. Per-batch
// s2+bias and z tables live in LDS (16 KB). The 16x64 P tile is double
// buffered: one barrier per round covers two K=32 WMMAs per wave. ELU fused.
// ---------------------------------------------------------------------------
__global__ __launch_bounds__(256)
void k_attn_gemm(const float* __restrict__ adj, const _Float16* __restrict__ hT,
                 const float* __restrict__ s1, const float* __restrict__ s2,
                 const float* __restrict__ zcol, const float* __restrict__ ab,
                 float* __restrict__ out) {
  __shared__ __align__(16) _Float16 sP[2][16 * 64];
  __shared__ float sS1[16];
  __shared__ float sS2b[NN];   // s2 + bias for this batch
  __shared__ float sZ[NN];     // max + log(denom) for this batch

  int blk  = blockIdx.x;            // B * N/16 = 1024
  int b    = blk >> 7;
  int i0   = (blk & 127) << 4;
  int t    = threadIdx.x;
  int lane = t & 31;
  int wave = t >> 5;
  int nw   = wave << 4;
  int bN   = b << 11;
  float bias = ab[0];

  if (t < 16) sS1[t] = s1[bN + i0 + t];
  for (int j = t; j < NN; j += 256) {
    sS2b[j] = s2[bN + j] + bias;
    sZ[j]   = zcol[bN + j];
  }
  __syncthreads();

  const float* adjb = adj + (size_t)bN * NN;
  const int o  = nw + (lane & 15);
  const _Float16* hrow = hT + (size_t)(b * FOUT + o) * NN;

  v8f c = {};
  for (int j0 = 0, it = 0; j0 < NN; j0 += 64, ++it) {
    _Float16* buf = sP[it & 1];
    // build probability tile P[16][64] (4 elements per thread, coalesced adj)
#pragma unroll
    for (int h = 0; h < 4; ++h) {
      int idx = t + h * 256;
      int il  = idx >> 6;
      int jj  = idx & 63;
      int j   = j0 + jj;
      const float* ap = &adjb[(size_t)(i0 + il) * NN + j];
      if (h == 0 && j0 + 64 < NN) __builtin_prefetch(ap + 64, 0, 1);
      float a = *ap;
      _Float16 p = (_Float16)0.f;
      if (a > 0.f)
        p = (_Float16)__expf(leaky(sS1[il] + sS2b[j]) - sZ[j]);
      buf[il * 64 + jj] = p;
    }
    __syncthreads();   // single barrier per round (ping-pong buffers)

    // two K=32 WMMA steps over the 16x64 tile
#pragma unroll
    for (int half = 0; half < 2; ++half) {
      int kb = half * 32;
      v16h a = load_A_from_lds(buf, 64, kb, lane);
      int jb = j0 + kb + ((lane < 16) ? 0 : 16);
      const v8h* hp = (const v8h*)(hrow + jb);
      v8h bl = hp[0], bh = hp[1];
      v16h bm;
#pragma unroll
      for (int p = 0; p < 8; ++p) { bm[p] = bl[p]; bm[p + 8] = bh[p]; }
      c = __builtin_amdgcn_wmma_f32_16x16x32_f16(false, a, false, bm,
                                                 (short)0, c, false, false);
    }
  }

  // ELU + store f32 output [b, i, o]
  int nl = lane & 15;
  int mh = (lane < 16) ? 0 : 8;
#pragma unroll
  for (int r = 0; r < 8; ++r) {
    float v = c[r];
    float ev = v > 0.f ? v : (__expf(v) - 1.f);
    out[(size_t)(bN + i0 + r + mh) * FOUT + nw + nl] = ev;
  }
}

// ---------------------------------------------------------------------------
extern "C" void kernel_launch(void* const* d_in, const int* in_sizes, int n_in,
                              void* d_out, int out_size, void* d_ws, size_t ws_size,
                              hipStream_t stream) {
  const float* x   = (const float*)d_in[0];   // [B,N,FIN]
  const float* adj = (const float*)d_in[1];   // [B,N,N]
  const float* W   = (const float*)d_in[2];   // [FOUT,FIN]
  const float* a1  = (const float*)d_in[3];   // [FOUT]
  const float* a2  = (const float*)d_in[4];   // [FOUT]
  const float* ab  = (const float*)d_in[5];   // [1]
  float* out = (float*)d_out;                 // [B,N,FOUT]

  // workspace carve-up
  char* ws = (char*)d_ws;
  _Float16* hT  = (_Float16*)ws;                               // 4 MB
  size_t off = (size_t)BB * FOUT * NN * sizeof(_Float16);
  float* s1   = (float*)(ws + off); off += (size_t)BB * NN * sizeof(float);
  float* s2   = (float*)(ws + off); off += (size_t)BB * NN * sizeof(float);
  float* zcol = (float*)(ws + off); off += (size_t)BB * NN * sizeof(float);
  float* pm   = (float*)(ws + off); off += (size_t)SEGS * BB * NN * sizeof(float);
  float* ps   = (float*)(ws + off);

  const int mblocks = BB * (NN / 16);   // 1024
  const int jblocks = BB * NN / 256;    // 64

  k_h_gemm<<<mblocks, 256, 0, stream>>>(x, W, hT);
  k_rowvec<<<jblocks, 256, 0, stream>>>(hT, a1, a2, s1, s2);
  k_colstats_partial<<<dim3(jblocks, SEGS), 256, 0, stream>>>(adj, s1, s2, ab, pm, ps);
  k_colstats_merge<<<jblocks, 256, 0, stream>>>(pm, ps, zcol);
  k_attn_gemm<<<mblocks, 256, 0, stream>>>(adj, hT, s1, s2, zcol, ab, out);
}
